// MultiheadGEAttention_45638322487491
// MI455X (gfx1250) — matrix operand order repaired and checked by reading, not compiled, not run on hardware
//
#include <hip/hip_runtime.h>
#include <hip/hip_bf16.h>

// ---------------------------------------------------------------------------
// MultiheadGEAttention for MI455X (gfx1250), wave32 + v_wmma_f32_16x16x32_f16
// ---------------------------------------------------------------------------

typedef __attribute__((ext_vector_type(16))) _Float16 v16h;
typedef __attribute__((ext_vector_type(8)))  _Float16 v8h;
typedef __attribute__((ext_vector_type(8)))  float    v8f;

#define BB   256          // batch
#define NN   129          // n_cols
#define DD   512          // d
#define HH   8            // heads
#define DHH  64           // d_head
#define NP   144          // padded N (9*16) for fh/ft rows and score cols
#define NK   160          // padded K (5*32) for the P@V GEMM
#define DCOL 15
#define BN   (BB*NN)      // 33024 (divisible by 16)
#define RTIL (BN/16)      // 2064 row tiles
#define CTIL (DD/16)      // 32 col tiles
#define CB   4            // col tiles per wave in projection GEMMs
#define CTB  (CTIL/CB)    // 8 col blocks

union F8 { v8f v; float f[8]; };

// A fragment: 16x32 f16 tile, rows contiguous over K (lda in halfs).
// ISA layout: lane L<16 holds row M=L, K in {k0+0..7} U {k0+16..23};
//             lane L>=16 holds row M=L-16, K in {k0+8..15} U {k0+24..31}.
__device__ __forceinline__ v16h frag_a(const _Float16* p0, int ld, int lane) {
    int r  = lane & 15;
    int hi = (lane >> 4) & 1;
    const _Float16* p = p0 + (size_t)r * ld + hi * 8;
    v8h lo = *(const v8h*)(p);
    v8h hh = *(const v8h*)(p + 16);
    return __builtin_shufflevector(lo, hh, 0,1,2,3,4,5,6,7,8,9,10,11,12,13,14,15);
}

// B fragment: 32x16 f16 tile; source stored as rows-per-output-column,
// contiguous over K (so B[k][n] = src[n][k]).  Lane L<16 -> col n=L, K in {k0+0..7} U {k0+16..23};
// lane L>=16 -> col n=L-16, K in {k0+8..15} U {k0+24..31}.
__device__ __forceinline__ v16h frag_b(const _Float16* p0, int ld, int lane) {
    int n  = lane & 15;
    int hi = (lane >> 4) & 1;
    const _Float16* p = p0 + (size_t)n * ld + hi * 8;
    v8h lo = *(const v8h*)(p);
    v8h hh = *(const v8h*)(p + 16);
    return __builtin_shufflevector(lo, hh, 0,1,2,3,4,5,6,7,8,9,10,11,12,13,14,15);
}

__device__ __forceinline__ v8f wmma_f16(v16h a, v16h b, v8f c) {
    return __builtin_amdgcn_wmma_f32_16x16x32_f16(false, a, false, b,
                                                  (short)0, c, false, false);
}

// ---------------------------------------------------------------------------
__global__ void zero_ws(unsigned int* __restrict__ p, long long n) {
    long long i = (long long)blockIdx.x * blockDim.x + threadIdx.x;
    long long st = (long long)gridDim.x * blockDim.x;
    for (; i < n; i += st) p[i] = 0u;
}

__global__ void f32_to_f16(const float* __restrict__ s, _Float16* __restrict__ d,
                           long long n) {
    long long i = (long long)blockIdx.x * blockDim.x + threadIdx.x;
    long long st = (long long)gridDim.x * blockDim.x;
    for (; i < n; i += st) d[i] = (_Float16)s[i];
}

// ---------------------------------------------------------------------------
// Topology mask: mask[h][r][c] = 0 if edge kept, -10000 if pruned, -1e9 in pads.
// sigmoid(cos + bias) > 0.5  <=>  cos + bias > 0.
__global__ void topo_mask(const float* __restrict__ ch, const float* __restrict__ ct,
                          const float* __restrict__ bias, float* __restrict__ mask) {
    int idx = blockIdx.x * blockDim.x + threadIdx.x;
    if (idx >= HH * NP * NP) return;
    int h = idx / (NP * NP);
    int r = (idx / NP) % NP;
    int c = idx % NP;
    float m = -1.0e9f;
    if (r < NN && c < NN) {
        const float* a = ch + ((size_t)h * NN + r) * DCOL;
        const float* b = ct + ((size_t)h * NN + c) * DCOL;
        float dot = 0.f, na = 0.f, nb = 0.f;
        #pragma unroll
        for (int i = 0; i < DCOL; ++i) {
            dot += a[i] * b[i]; na += a[i] * a[i]; nb += b[i] * b[i];
        }
        float cosv = dot / (fmaxf(sqrtf(na), 1e-12f) * fmaxf(sqrtf(nb), 1e-12f));
        bool adj = (cosv + bias[0] > 0.0f) && (r != c) && (c != 0);
        m = adj ? 0.0f : -10000.0f;
    }
    mask[idx] = m;
}

// ---------------------------------------------------------------------------
// Fused QKV projections. One wave = 16x64 output strip (4 col tiles, A reused 4x).
// which 0: f_head = xh @ Wh^T (+b) scaled by rel_emb/8 -> fh [b,h,NP,64] f16
// which 1: f_tail = xt @ Wh^T (+b)                     -> ft [b,h,NP,64] f16
// which 2: f_v    = xt @ Wv^T (+b)  transposed store   -> fvT[b,h,64,NK] f16
__global__ __launch_bounds__(256) void qkv_gemm(
    const _Float16* __restrict__ xh, const _Float16* __restrict__ xt,
    const _Float16* __restrict__ wh, const _Float16* __restrict__ wv,
    const float* __restrict__ relemb, const float* __restrict__ bh_,
    const float* __restrict__ bv_,
    _Float16* __restrict__ fh, _Float16* __restrict__ ft,
    _Float16* __restrict__ fvT) {
    int lane = threadIdx.x & 31;
    int wave = threadIdx.x >> 5;
    int tile  = blockIdx.x * 8 + wave;            // 3 * RTIL * CTB total
    int which = tile / (RTIL * CTB);
    int t2    = tile % (RTIL * CTB);
    int rt = t2 / CTB, ctb = t2 % CTB;

    const _Float16* A  = ((which == 0) ? xh : xt) + (size_t)rt * 16 * DD;
    const _Float16* Bm = ((which == 2) ? wv : wh) + (size_t)ctb * (CB * 16) * DD;

    F8 acc[CB] = {};
    for (int k = 0; k < DD; k += 32) {
        v16h a = frag_a(A + k, DD, lane);
        #pragma unroll
        for (int cb = 0; cb < CB; ++cb) {
            v16h b = frag_b(Bm + (size_t)cb * 16 * DD + k, DD, lane);
            acc[cb].v = wmma_f16(a, b, acc[cb].v);
        }
    }

    int hi = (lane >> 4) & 1;
    #pragma unroll
    for (int cb = 0; cb < CB; ++cb) {
        int e  = (ctb * CB + cb) * 16 + (lane & 15);  // output feature
        int h  = e >> 6, dh = e & 63;                 // head / within-head
        #pragma unroll
        for (int j = 0; j < 8; ++j) {
            int row = rt * 16 + j + hi * 8;           // flat (b,n)
            int b_ = row / NN, n = row % NN;
            float v = acc[cb].f[j];
            if (which == 0) {
                v = (v + bh_[e]) * relemb[e] * 0.125f;  // rel_emb and 1/sqrt(64)
                fh[(((size_t)b_ * HH + h) * NP + n) * DHH + dh] = (_Float16)v;
            } else if (which == 1) {
                v += bh_[e];
                ft[(((size_t)b_ * HH + h) * NP + n) * DHH + dh] = (_Float16)v;
            } else {
                v += bv_[e];
                fvT[(((size_t)b_ * HH + h) * DHH + dh) * NK + n] = (_Float16)v;
            }
        }
    }
}

// ---------------------------------------------------------------------------
// Attention: one block per (b, h, row-tile). 9 waves, 288 threads.
// Phase 1: wave w computes S[16x16] col-tile w via WMMA (K=64), adds mask -> LDS.
// Phase 2: parallel f32 softmax: wave w (w<8) handles rows 2w, 2w+1; each
//          16-lane half-wave strides the 144 cols, reduces via __shfl_xor.
// Phase 3: write fr_graph (f32); waves 0..3 compute O = P @ fv via WMMA (K=160).
__global__ __launch_bounds__(288) void attention(
    const _Float16* __restrict__ fh, const _Float16* __restrict__ ft,
    const _Float16* __restrict__ fvT, const float* __restrict__ mask,
    _Float16* __restrict__ xa, float* __restrict__ fr) {
    __shared__ __align__(16) float    S[16][NK];
    __shared__ __align__(16) _Float16 P[16][NK];

    int tid = threadIdx.x, lane = tid & 31, wave = tid >> 5;
    int hi = (lane >> 4) & 1;
    int rt = blockIdx.x % 9;
    int bh = blockIdx.x / 9;
    int b_ = bh / HH, h = bh % HH;

    {   // scores
        int ct = wave;
        F8 acc = {};
        const _Float16* Ab = fh + ((size_t)bh * NP + rt * 16) * DHH;
        const _Float16* Bb = ft + ((size_t)bh * NP + ct * 16) * DHH;
        #pragma unroll
        for (int k = 0; k < DHH; k += 32) {
            v16h a = frag_a(Ab + k, DHH, lane);
            v16h b = frag_b(Bb + k, DHH, lane);
            acc.v = wmma_f16(a, b, acc.v);
        }
        int cc = ct * 16 + (lane & 15);
        #pragma unroll
        for (int j = 0; j < 8; ++j) {
            int r  = j + hi * 8;
            int gr = rt * 16 + r;
            S[r][cc] = acc.f[j] + mask[((size_t)h * NP + gr) * NP + cc];
        }
    }
    __syncthreads();

    if (wave < 8) {                     // softmax: half-wave per row
        int row = wave * 2 + hi;
        int l   = lane & 15;
        float m = -1.0e30f;
        for (int c = l; c < NP; c += 16) m = fmaxf(m, S[row][c]);
        #pragma unroll
        for (int off = 1; off < 16; off <<= 1)
            m = fmaxf(m, __shfl_xor(m, off, 16));
        float s = 0.f;
        for (int c = l; c < NP; c += 16) {
            float e = __expf(S[row][c] - m);
            S[row][c] = e;
            s += e;
        }
        #pragma unroll
        for (int off = 1; off < 16; off <<= 1)
            s += __shfl_xor(s, off, 16);
        float inv = (s > 0.f) ? 1.0f / s : 0.0f;
        for (int c = l; c < NK; c += 16) {
            float p = (c < NP) ? S[row][c] * inv : 0.0f;
            if (c < NP) S[row][c] = p;
            P[row][c] = (_Float16)p;
        }
    }
    __syncthreads();

    for (int idx = tid; idx < 16 * NN; idx += 288) {     // fr_graph out (f32)
        int r = idx / NN, c = idx % NN;
        int n = rt * 16 + r;
        if (n < NN)
            fr[(((size_t)b_ * HH + h) * NN + n) * NN + c] = S[r][c];
    }

    if (wave < 4) {                                      // O = P @ fv
        int dt = wave;
        F8 acc = {};
        const _Float16* Bb = fvT + ((size_t)bh * DHH + dt * 16) * NK;
        #pragma unroll
        for (int ks = 0; ks < NK; ks += 32) {
            v16h a = frag_a(&P[0][0] + ks, NK, lane);
            v16h b = frag_b(Bb + ks, NK, lane);
            acc.v = wmma_f16(a, b, acc.v);
        }
        int d = h * DHH + dt * 16 + (lane & 15);
        #pragma unroll
        for (int j = 0; j < 8; ++j) {
            int n = rt * 16 + j + hi * 8;
            if (n < NN)
                xa[((size_t)b_ * NN + n) * DD + d] = (_Float16)acc.f[j];
        }
    }
}

// ---------------------------------------------------------------------------
// Output projection: one wave = 16x64 strip (4 col tiles, A reused 4x).
__global__ __launch_bounds__(256) void out_gemm(
    const _Float16* __restrict__ xa, const _Float16* __restrict__ wo,
    const float* __restrict__ bo, float* __restrict__ out) {
    int lane = threadIdx.x & 31;
    int wave = threadIdx.x >> 5;
    int tile = blockIdx.x * 8 + wave;            // RTIL * CTB total
    int rt = tile / CTB, ctb = tile % CTB;
    const _Float16* A  = xa + (size_t)rt * 16 * DD;
    const _Float16* Bm = wo + (size_t)ctb * (CB * 16) * DD;

    F8 acc[CB] = {};
    for (int k = 0; k < DD; k += 32) {
        v16h a = frag_a(A + k, DD, lane);
        #pragma unroll
        for (int cb = 0; cb < CB; ++cb) {
            v16h b = frag_b(Bm + (size_t)cb * 16 * DD + k, DD, lane);
            acc[cb].v = wmma_f16(a, b, acc[cb].v);
        }
    }

    int hi = (lane >> 4) & 1;
    #pragma unroll
    for (int cb = 0; cb < CB; ++cb) {
        int e = (ctb * CB + cb) * 16 + (lane & 15);
        float be = bo[e];
        #pragma unroll
        for (int j = 0; j < 8; ++j) {
            int row = rt * 16 + j + hi * 8;
            out[(size_t)row * DD + e] = acc[cb].f[j] + be;
        }
    }
}

// ---------------------------------------------------------------------------
extern "C" void kernel_launch(void* const* d_in, const int* in_sizes, int n_in,
                              void* d_out, int out_size, void* d_ws, size_t ws_size,
                              hipStream_t stream) {
    (void)in_sizes; (void)n_in; (void)out_size; (void)ws_size;
    const float* x_head   = (const float*)d_in[0];
    const float* x_tail   = (const float*)d_in[1];
    const float* W_head_w = (const float*)d_in[2];
    const float* W_head_b = (const float*)d_in[3];
    const float* W_v_w    = (const float*)d_in[4];
    const float* W_v_b    = (const float*)d_in[5];
    const float* W_out_w  = (const float*)d_in[6];
    const float* W_out_b  = (const float*)d_in[7];
    const float* rel_emb  = (const float*)d_in[8];
    const float* col_head = (const float*)d_in[9];
    const float* col_tail = (const float*)d_in[10];
    const float* bias     = (const float*)d_in[11];

    char* ws = (char*)d_ws;
    size_t off = 0;
    _Float16* xh16 = (_Float16*)(ws + off); off += (size_t)BN * DD * 2;
    _Float16* xt16 = (_Float16*)(ws + off); off += (size_t)BN * DD * 2;
    _Float16* wh16 = (_Float16*)(ws + off); off += (size_t)DD * DD * 2;
    _Float16* wv16 = (_Float16*)(ws + off); off += (size_t)DD * DD * 2;
    _Float16* wo16 = (_Float16*)(ws + off); off += (size_t)DD * DD * 2;
    _Float16* fh   = (_Float16*)(ws + off); off += (size_t)BB * HH * NP * DHH * 2;
    _Float16* ft   = (_Float16*)(ws + off); off += (size_t)BB * HH * NP * DHH * 2;
    _Float16* fvT  = (_Float16*)(ws + off); off += (size_t)BB * HH * DHH * NK * 2;
    float*    adjm = (float*)   (ws + off); off += (size_t)HH * NP * NP * 4;

    // fh, ft, fvT are contiguous: zero once so all pads are exactly 0.
    long long zwords = ((long long)BB * HH * NP * DHH * 2 * 2 +
                        (long long)BB * HH * DHH * NK * 2) / 4;
    zero_ws<<<4096, 256, 0, stream>>>((unsigned int*)fh, zwords);

    f32_to_f16<<<4096, 256, 0, stream>>>(x_head,   xh16, (long long)BN * DD);
    f32_to_f16<<<4096, 256, 0, stream>>>(x_tail,   xt16, (long long)BN * DD);
    f32_to_f16<<<512,  256, 0, stream>>>(W_head_w, wh16, (long long)DD * DD);
    f32_to_f16<<<512,  256, 0, stream>>>(W_v_w,    wv16, (long long)DD * DD);
    f32_to_f16<<<512,  256, 0, stream>>>(W_out_w,  wo16, (long long)DD * DD);

    topo_mask<<<(HH * NP * NP + 255) / 256, 256, 0, stream>>>(col_head, col_tail,
                                                              bias, adjm);

    qkv_gemm<<<(3 * RTIL * CTB) / 8, 256, 0, stream>>>(
        xh16, xt16, wh16, wv16, rel_emb, W_head_b, W_v_b, fh, ft, fvT);

    float* out_x = (float*)d_out;
    float* fr    = out_x + (size_t)BN * DD;          // fr_graph after x
    _Float16* xa16 = xh16;                           // reuse: xh16 dead now

    attention<<<BB * HH * 9, 288, 0, stream>>>(fh, ft, fvT, adjm, xa16, fr);

    out_gemm<<<(RTIL * CTB) / 8, 256, 0, stream>>>(xa16, wo16, W_out_b, out_x);
}
